// MultiHeadAttention_47055661695479
// MI455X (gfx1250) — compile-verified, matrix operand order checked by reading, not run on hardware
//
#include <hip/hip_runtime.h>

// ---------------- types ----------------
typedef __bf16 bf16;
typedef __attribute__((ext_vector_type(2)))  __bf16 v2bf;
typedef __attribute__((ext_vector_type(4)))  __bf16 v4bf;
typedef __attribute__((ext_vector_type(8)))  __bf16 v8bf;
typedef __attribute__((ext_vector_type(16))) __bf16 v16bf;
typedef __attribute__((ext_vector_type(8)))  float  v8f;
typedef __attribute__((ext_vector_type(4)))  unsigned int v4u;
typedef __attribute__((ext_vector_type(8)))  int v8i;
typedef __attribute__((ext_vector_type(4)))  int v4i;

static constexpr int B_ = 4, T_ = 2048, D_ = 1024, H_ = 16, DH_ = 64;

__device__ __forceinline__ v8f wmma_bf16(v16bf a, v16bf b, v8f c) {
  return __builtin_amdgcn_wmma_f32_16x16x32_bf16(false, a, false, b, (short)0, c,
                                                 false, false);
}

// 16x32 bf16 A-fragment from row-major storage (row stride = `stride` elems),
// or 32x16 B-fragment from column-contiguous (K-major) storage.
__device__ __forceinline__ v16bf load_frag(const bf16* base, int stride, int lane) {
  const int m  = lane & 15;
  const int kb = (lane & 16) >> 1;  // 0 or 8
  const bf16* p = base + (size_t)m * stride + kb;
  v8bf lo = *(const v8bf*)(p);
  v8bf hi = *(const v8bf*)(p + 16);
  return __builtin_shufflevector(lo, hi, 0, 1, 2, 3, 4, 5, 6, 7, 8, 9, 10, 11,
                                 12, 13, 14, 15);
}

// 2-D TDM tile load (bf16) per CDNA5 ISA.
// Group0: count=1, lds_addr, 57b global byte address, type=2.
// Group1: data_size=2B, tensor dim0/dim1, tile dim0/dim1, dim0 stride.
// This toolchain's builtin: (g0 v4u, g1 v8i, g2 v4i, g3 v4i, g4 v8i, cpol).
__device__ __forceinline__ void tdm_load_2d(unsigned lds_off, const bf16* gptr,
                                            unsigned tensor_d0, unsigned tensor_d1,
                                            unsigned stride0, unsigned tile_d0,
                                            unsigned tile_d1) {
  const unsigned long long ga = (unsigned long long)(uintptr_t)gptr;
  v4u g0;
  g0[0] = 1u;                                            // count=1 (valid user D#)
  g0[1] = lds_off;                                       // LDS byte address
  g0[2] = (unsigned)(ga & 0xffffffffu);                  // global_addr[31:0]
  g0[3] = (unsigned)((ga >> 32) & 0x01ffffffu) | (2u << 30);  // addr[56:32]|type=2
  v8i g1;
  g1[0] = (int)(1u << 16);                               // data_size=1 -> 2 bytes
  g1[1] = (int)((tensor_d0 & 0xffffu) << 16);
  g1[2] = (int)(((tensor_d0 >> 16) & 0xffffu) | ((tensor_d1 & 0xffffu) << 16));
  g1[3] = (int)(((tensor_d1 >> 16) & 0xffffu) | ((tile_d0 & 0xffffu) << 16));
  g1[4] = (int)(tile_d1 & 0xffffu);                      // tile1 | tile2=0
  g1[5] = (int)stride0;                                  // dim0_stride[31:0]
  g1[6] = 0;
  g1[7] = 0;
  const v4i z4 = {0, 0, 0, 0};
  const v8i z8 = {0, 0, 0, 0, 0, 0, 0, 0};
  __builtin_amdgcn_tensor_load_to_lds(g0, g1, z4, z4, z8, 0);
}

// ---------------- fp32 -> bf16 convert ----------------
__global__ void cvt_f32_bf16_kernel(const float* __restrict__ src,
                                    bf16* __restrict__ dst, int n4) {
  int i = blockIdx.x * blockDim.x + threadIdx.x;
  if (i < n4) {
    const float4 v = ((const float4*)src)[i];
    v4bf o;
    o[0] = (bf16)v.x; o[1] = (bf16)v.y; o[2] = (bf16)v.z; o[3] = (bf16)v.w;
    ((v4bf*)dst)[i] = o;
  }
}

// ---------------- QKV projection ----------------
// 4 waves/block share (proj,b,h) => identical weight fragments.  A leader wave
// TDM-stages 64x128 W slabs into LDS (double-buffered); waves read B-fragments
// from LDS and stream their own A rows from global (pipelined, no reg copies).
__global__ __launch_bounds__(128) void qkv_proj_kernel(
    const bf16* __restrict__ xb, const bf16* __restrict__ wq,
    const bf16* __restrict__ wk, const bf16* __restrict__ wv,
    bf16* __restrict__ Q, bf16* __restrict__ K, bf16* __restrict__ Vt) {
  __shared__ __align__(16) bf16 wlds[2][64 * 128];
  const int lane = threadIdx.x & 31;
  const int wid  = blockIdx.x * 4 + (threadIdx.x >> 5);
  const int tT   = wid & 127;
  const int h    = (wid >> 7) & 15;
  const int b    = (wid >> 11) & 3;
  const int proj = wid >> 13;  // 0=Q 1=K 2=V

  const bf16* W = (proj == 0 ? wq : proj == 1 ? wk : wv) + (size_t)h * DH_ * D_;
  const bf16* A = xb + ((size_t)b * T_ + tT * 16) * D_;

  const int wslot = __builtin_amdgcn_readfirstlane(threadIdx.x >> 5);
  if (wslot == 0)  // slab 0: W cols 0..127 of all 64 out-rows
    tdm_load_2d((unsigned)(uintptr_t)&wlds[0][0], W, D_, DH_, D_, 128, DH_);

  v8f acc[4] = {};
  v16bf af[2];
  af[0] = load_frag(A, D_, lane);

  for (int s = 0; s < 8; ++s) {  // 8 slabs of 128 K
    const int cur = s & 1;
    if (wslot == 0) {
      if (s + 1 < 8) {
        tdm_load_2d((unsigned)(uintptr_t)&wlds[cur ^ 1][0], W + (s + 1) * 128,
                    D_, DH_, D_, 128, DH_);
        __builtin_amdgcn_s_wait_tensorcnt(1);  // current slab done, next in flight
      } else {
        __builtin_amdgcn_s_wait_tensorcnt(0);
      }
    }
    __syncthreads();
    const bf16* wp = &wlds[cur][0];  // [64 out][128 k], stride 128
#pragma unroll
    for (int i = 0; i < 4; ++i) {    // 4 k-steps of 32 per slab
      const int kg = s * 128 + i * 32;
      if (kg + 32 < D_)  // prefetch next A fragment (distinct reg, no copies)
        af[(i + 1) & 1] = load_frag(A + kg + 32, D_, lane);
#pragma unroll
      for (int c = 0; c < 4; ++c) {
        v16bf bw = load_frag(wp + (size_t)(c * 16) * 128 + i * 32, 128, lane);
        acc[c] = wmma_bf16(af[i & 1], bw, acc[c]);
      }
    }
    __syncthreads();  // all waves done with wlds[cur] before leader overwrites
  }

  const int half = lane >> 4, N = lane & 15;
  if (proj == 2) {
    // V stored transposed: Vt[b,h,e,t] -> 8 consecutive t's per lane = one b128.
#pragma unroll
    for (int c = 0; c < 4; ++c) {
      v8bf pack;
#pragma unroll
      for (int r = 0; r < 8; ++r) pack[r] = (bf16)acc[c][r];
      bf16* dst = Vt + ((size_t)(b * H_ + h) * DH_ + c * 16 + N) * T_ +
                  tT * 16 + half * 8;
      *(v8bf*)dst = pack;
    }
  } else {
    bf16* O = (proj == 0 ? Q : K) + ((size_t)(b * H_ + h) * T_ + tT * 16) * DH_;
#pragma unroll
    for (int c = 0; c < 4; ++c)
#pragma unroll
      for (int r = 0; r < 8; ++r)
        O[(size_t)(r + half * 8) * DH_ + c * 16 + N] = (bf16)acc[c][r];
  }
}

// ---------------- flash attention ----------------
// Leader wave TDM-stages the shared 32-key K/V tiles (double-buffered); the
// j-loop is unrolled x2 so buffer addresses are compile-time constants.  P is
// packed two-per-u32 in LDS and read back as an A-fragment with a deinterleave.
__global__ __launch_bounds__(128) void flash_attn_kernel(
    const bf16* __restrict__ Q, const bf16* __restrict__ K,
    const bf16* __restrict__ Vt, bf16* __restrict__ ctx) {
  __shared__ __align__(16) unsigned pbuf[4][16 * 16];
  __shared__ __align__(16) bf16 kbuf[2][32 * 64];
  __shared__ __align__(16) bf16 vbuf[2][64 * 32];
  const int lane = threadIdx.x & 31;
  const int ws   = threadIdx.x >> 5;
  const int wid  = blockIdx.x * 4 + ws;
  const int qT   = wid & 127;
  const int h    = (wid >> 7) & 15;
  const int b    = (wid >> 11) & 3;

  const bf16* Qb = Q + ((size_t)(b * H_ + h) * T_ + qT * 16) * DH_;
  const bf16* Kb = K + (size_t)(b * H_ + h) * T_ * DH_;
  const bf16* Vb = Vt + (size_t)(b * H_ + h) * DH_ * T_;

  const v16bf qf0 = load_frag(Qb + 0, DH_, lane);
  const v16bf qf1 = load_frag(Qb + 32, DH_, lane);

  v8f o[4] = {};
  float m[8], l[8];
#pragma unroll
  for (int r = 0; r < 8; ++r) { m[r] = -3.0e38f; l[r] = 0.0f; }
  const float scale = 0.125f;  // 1/sqrt(DH)
  const int half = lane >> 4, N = lane & 15;

  const int wslot = __builtin_amdgcn_readfirstlane(threadIdx.x >> 5);
  auto tdm_tiles = [&](int jj, int buf) {
    tdm_load_2d((unsigned)(uintptr_t)&kbuf[buf][0], Kb + (size_t)jj * DH_,
                DH_, T_, DH_, DH_, 32);
    tdm_load_2d((unsigned)(uintptr_t)&vbuf[buf][0], Vb + jj,
                T_, DH_, T_, 32, DH_);
  };

  // One 32-key step against staged LDS tiles.
  auto tile_step = [&](const bf16* kt, const bf16* vt) {
    v16bf k00 = load_frag(kt + 0 * 64 + 0, 64, lane);
    v16bf k01 = load_frag(kt + 0 * 64 + 32, 64, lane);
    v16bf k10 = load_frag(kt + 16 * 64 + 0, 64, lane);
    v16bf k11 = load_frag(kt + 16 * 64 + 32, 64, lane);
    v8f s0 = {}, s1 = {};
    s0 = wmma_bf16(qf0, k00, s0);
    s0 = wmma_bf16(qf1, k01, s0);
    s1 = wmma_bf16(qf0, k10, s1);
    s1 = wmma_bf16(qf1, k11, s1);

    float fac[8];
#pragma unroll
    for (int r = 0; r < 8; ++r) {
      float a0 = s0[r] * scale, a1 = s1[r] * scale;
      float mx = fmaxf(a0, a1);
      mx = fmaxf(mx, __shfl_xor(mx, 1, 32));
      mx = fmaxf(mx, __shfl_xor(mx, 2, 32));
      mx = fmaxf(mx, __shfl_xor(mx, 4, 32));
      mx = fmaxf(mx, __shfl_xor(mx, 8, 32));
      const float mn = fmaxf(m[r], mx);
      const float f  = __expf(m[r] - mn);
      const float p0 = __expf(a0 - mn);
      const float p1 = __expf(a1 - mn);
      float rs = p0 + p1;
      rs += __shfl_xor(rs, 1, 32);
      rs += __shfl_xor(rs, 2, 32);
      rs += __shfl_xor(rs, 4, 32);
      rs += __shfl_xor(rs, 8, 32);
      l[r] = l[r] * f + rs;
      m[r] = mn;
      fac[r] = f;
      // pack P(K=N) | P(K=N+16) into one u32 LDS word
      const unsigned w = ((unsigned)__builtin_bit_cast(unsigned short, (bf16)p1) << 16) |
                         (unsigned)__builtin_bit_cast(unsigned short, (bf16)p0);
      pbuf[ws][(r + half * 8) * 16 + N] = w;
    }
#pragma unroll
    for (int c = 0; c < 4; ++c)
#pragma unroll
      for (int r = 0; r < 8; ++r) o[c][r] *= fac[r];

    // Read P back as an A-fragment: words kb..kb+7 of row m hold lo=K=kb+i,
    // hi=K=kb+16+i -> two b128 loads + one deinterleave shuffle.
    const unsigned* prow = &pbuf[ws][(lane & 15) * 16 + ((lane & 16) >> 1)];
    v8bf plo = *(const v8bf*)(prow);
    v8bf phi = *(const v8bf*)(prow + 4);
    v16bf inter = __builtin_shufflevector(plo, phi, 0, 1, 2, 3, 4, 5, 6, 7, 8, 9,
                                          10, 11, 12, 13, 14, 15);
    v16bf pf = __builtin_shufflevector(inter, inter, 0, 2, 4, 6, 8, 10, 12, 14,
                                       1, 3, 5, 7, 9, 11, 13, 15);
#pragma unroll
    for (int c = 0; c < 4; ++c) {
      v16bf vf = load_frag(vt + (size_t)(c * 16) * 32, 32, lane);
      o[c] = wmma_bf16(pf, vf, o[c]);
    }
  };

  if (wslot == 0) tdm_tiles(0, 0);

  for (int j0 = 0; j0 < T_; j0 += 64) {
    // ---- stage 0: keys j0, buffer 0 ----
    if (wslot == 0) {
      tdm_tiles(j0 + 32, 1);                 // always valid: j0 <= T-64
      __builtin_amdgcn_s_wait_tensorcnt(2);  // buf0 ready, buf1 in flight
    }
    __syncthreads();
    tile_step(&kbuf[0][0], &vbuf[0][0]);
    __syncthreads();
    // ---- stage 1: keys j0+32, buffer 1 ----
    if (wslot == 0) {
      if (j0 + 64 < T_) {
        tdm_tiles(j0 + 64, 0);
        __builtin_amdgcn_s_wait_tensorcnt(2);
      } else {
        __builtin_amdgcn_s_wait_tensorcnt(0);
      }
    }
    __syncthreads();
    tile_step(&kbuf[1][0], &vbuf[1][0]);
    __syncthreads();
  }

#pragma unroll
  for (int c = 0; c < 4; ++c)
#pragma unroll
    for (int r = 0; r < 8; ++r) {
      float val = o[c][r] / l[r];
      ctx[((size_t)b * T_ + qT * 16 + r + half * 8) * D_ + h * DH_ + c * 16 + N] =
          (bf16)val;
    }
}

// ---------------- output projection + bias ----------------
// Decode puts rT in the low bits: 4 waves/block share nT => shared W slabs,
// TDM-staged exactly like qkv_proj.
__global__ __launch_bounds__(128) void out_proj_kernel(
    const bf16* __restrict__ ctx, const bf16* __restrict__ wo,
    const float* __restrict__ bo, float* __restrict__ out) {
  __shared__ __align__(16) bf16 wlds[2][64 * 128];
  const int lane = threadIdx.x & 31;
  const int wid  = blockIdx.x * 4 + (threadIdx.x >> 5);
  const int rT   = wid & 511;  // 16-row tile of B*T
  const int nT   = wid >> 9;   // 64-wide output column tile

  const bf16* A = ctx + (size_t)rT * 16 * D_;
  const bf16* W = wo + (size_t)(nT * 64) * D_;

  const int wslot = __builtin_amdgcn_readfirstlane(threadIdx.x >> 5);
  if (wslot == 0)
    tdm_load_2d((unsigned)(uintptr_t)&wlds[0][0], W, D_, DH_, D_, 128, DH_);

  v8f acc[4] = {};
  v16bf af[2];
  af[0] = load_frag(A, D_, lane);

  for (int s = 0; s < 8; ++s) {
    const int cur = s & 1;
    if (wslot == 0) {
      if (s + 1 < 8) {
        tdm_load_2d((unsigned)(uintptr_t)&wlds[cur ^ 1][0], W + (s + 1) * 128,
                    D_, DH_, D_, 128, DH_);
        __builtin_amdgcn_s_wait_tensorcnt(1);
      } else {
        __builtin_amdgcn_s_wait_tensorcnt(0);
      }
    }
    __syncthreads();
    const bf16* wp = &wlds[cur][0];
#pragma unroll
    for (int i = 0; i < 4; ++i) {
      const int kg = s * 128 + i * 32;
      if (kg + 32 < D_)
        af[(i + 1) & 1] = load_frag(A + kg + 32, D_, lane);
#pragma unroll
      for (int c = 0; c < 4; ++c) {
        v16bf bw = load_frag(wp + (size_t)(c * 16) * 128 + i * 32, 128, lane);
        acc[c] = wmma_bf16(af[i & 1], bw, acc[c]);
      }
    }
    __syncthreads();
  }

  const int half = lane >> 4, N = lane & 15;
#pragma unroll
  for (int c = 0; c < 4; ++c) {
    const float bias = bo[nT * 64 + c * 16 + N];
#pragma unroll
    for (int r = 0; r < 8; ++r)
      out[(size_t)(rT * 16 + r + half * 8) * D_ + nT * 64 + c * 16 + N] =
          acc[c][r] + bias;
  }
}

// ---------------- launch ----------------
extern "C" void kernel_launch(void* const* d_in, const int* in_sizes, int n_in,
                              void* d_out, int out_size, void* d_ws,
                              size_t ws_size, hipStream_t stream) {
  (void)in_sizes; (void)n_in; (void)out_size; (void)ws_size;
  const float* x  = (const float*)d_in[0];
  const float* wq = (const float*)d_in[1];
  const float* wk = (const float*)d_in[2];
  const float* wv = (const float*)d_in[3];
  const float* wo = (const float*)d_in[4];
  const float* bo = (const float*)d_in[5];
  float* out = (float*)d_out;

  char* ws = (char*)d_ws;
  size_t off = 0;
  auto alloc = [&](size_t bytes) -> char* {
    char* p = ws + off;
    off += (bytes + 255) & ~(size_t)255;
    return p;
  };
  bf16* xb  = (bf16*)alloc((size_t)B_ * T_ * D_ * 2);
  bf16* wqb = (bf16*)alloc((size_t)H_ * DH_ * D_ * 2);
  bf16* wkb = (bf16*)alloc((size_t)H_ * DH_ * D_ * 2);
  bf16* wvb = (bf16*)alloc((size_t)H_ * DH_ * D_ * 2);
  bf16* wob = (bf16*)alloc((size_t)D_ * D_ * 2);
  bf16* Qd  = (bf16*)alloc((size_t)B_ * H_ * T_ * DH_ * 2);
  bf16* Kd  = (bf16*)alloc((size_t)B_ * H_ * T_ * DH_ * 2);
  bf16* Vtd = (bf16*)alloc((size_t)B_ * H_ * DH_ * T_ * 2);
  bf16* cxd = (bf16*)alloc((size_t)B_ * T_ * D_ * 2);

  cvt_f32_bf16_kernel<<<(B_ * T_ * D_ / 4) / 256, 256, 0, stream>>>(
      x, xb, B_ * T_ * D_ / 4);
  cvt_f32_bf16_kernel<<<(H_ * DH_ * D_ / 4) / 256, 256, 0, stream>>>(
      wq, wqb, H_ * DH_ * D_ / 4);
  cvt_f32_bf16_kernel<<<(H_ * DH_ * D_ / 4) / 256, 256, 0, stream>>>(
      wk, wkb, H_ * DH_ * D_ / 4);
  cvt_f32_bf16_kernel<<<(H_ * DH_ * D_ / 4) / 256, 256, 0, stream>>>(
      wv, wvb, H_ * DH_ * D_ / 4);
  cvt_f32_bf16_kernel<<<(D_ * D_ / 4) / 256, 256, 0, stream>>>(
      wo, wob, D_ * D_ / 4);

  qkv_proj_kernel<<<3 * B_ * H_ * (T_ / 16) / 4, 128, 0, stream>>>(
      xb, wqb, wkb, wvb, Qd, Kd, Vtd);
  flash_attn_kernel<<<B_ * H_ * (T_ / 16) / 4, 128, 0, stream>>>(Qd, Kd, Vtd,
                                                                 cxd);
  out_proj_kernel<<<(B_ * T_ / 16) * (D_ / 64) / 4, 128, 0, stream>>>(cxd, wob,
                                                                      bo, out);
}
